// TTLoRAGenerationWrapper_21938692948562
// MI455X (gfx1250) — matrix operand order repaired and checked by reading, not compiled, not run on hardware
//
#include <hip/hip_runtime.h>

typedef __attribute__((ext_vector_type(16))) __bf16          v16bf;
typedef __attribute__((ext_vector_type(8)))  float           v8f;
typedef __attribute__((ext_vector_type(4)))  float           v4f;
typedef __attribute__((ext_vector_type(16))) unsigned short  v16u;
typedef __attribute__((ext_vector_type(8)))  unsigned short  v8u;

#define D_IN   1024
#define D_OUT  1024
#define RANK   8
#define ALPHA  2.0f

__device__ __forceinline__ unsigned short f2bf(float f) {
  union { float f; unsigned u; } c; c.f = f;
  unsigned u = c.u;
  // round-to-nearest-even truncation f32 -> bf16
  return (unsigned short)((u + 0x7FFFu + ((u >> 16) & 1u)) >> 16);
}

__device__ __forceinline__ v16u cat8(v8u lo, v8u hi) {
  return __builtin_shufflevector(lo, hi, 0, 1, 2, 3, 4, 5, 6, 7,
                                 8, 9, 10, 11, 12, 13, 14, 15);
}

// ---------------------------------------------------------------------------
// Kernel 1: contract the six TT cores into the rank-8 factorization
//   A3 [in=1024][r=8]  (input-side chain:  core0*core1*core2)
//   BmT[out=1024][r=8] (output-side chain: core3*core4*core5, out = n5*64+n4*8+n3)
// ---------------------------------------------------------------------------
__global__ void tt_contract_cores(const float* __restrict__ c0, const float* __restrict__ c1,
                                  const float* __restrict__ c2, const float* __restrict__ c3,
                                  const float* __restrict__ c4, const float* __restrict__ c5,
                                  float* __restrict__ A3, float* __restrict__ BmT) {
  const int t = threadIdx.x;
  for (int in = t; in < D_IN; in += blockDim.x) {
    const int m1 = in >> 7, m2 = (in >> 4) & 7, m3 = in & 15;
    float tmp[RANK];
#pragma unroll
    for (int r2 = 0; r2 < RANK; ++r2) {
      float s = 0.f;
#pragma unroll
      for (int r1 = 0; r1 < RANK; ++r1)
        s += c0[m1 * 8 + r1] * c1[(r1 * 8 + m2) * 8 + r2];
      tmp[r2] = s;
    }
#pragma unroll
    for (int r3 = 0; r3 < RANK; ++r3) {
      float s = 0.f;
#pragma unroll
      for (int r2 = 0; r2 < RANK; ++r2)
        s += tmp[r2] * c2[(r2 * 16 + m3) * 8 + r3];
      A3[in * RANK + r3] = s;
    }
  }
  for (int out = t; out < D_OUT; out += blockDim.x) {
    const int n3 = out & 7, n4 = (out >> 3) & 7, n5 = out >> 6;
    float s4[RANK];
#pragma unroll
    for (int r4 = 0; r4 < RANK; ++r4) {
      float s = 0.f;
#pragma unroll
      for (int r5 = 0; r5 < RANK; ++r5)
        s += c4[(r4 * 8 + n4) * 8 + r5] * c5[r5 * 16 + n5];
      s4[r4] = s;
    }
#pragma unroll
    for (int r3 = 0; r3 < RANK; ++r3) {
      float s = 0.f;
#pragma unroll
      for (int r4 = 0; r4 < RANK; ++r4)
        s += c3[(r3 * 8 + n3) * 8 + r4] * s4[r4];
      BmT[out * RANK + r3] = s;
    }
  }
}

// ---------------------------------------------------------------------------
// Kernel 2: fuse base weight + alpha * (A3 x BmT), cast to bf16.
// Wc is n-major (Wc[n*1024+k]); w_base is [out][in] row-major so idx matches.
// ---------------------------------------------------------------------------
__global__ void build_fused_w(const float* __restrict__ wb, const float* __restrict__ A3,
                              const float* __restrict__ BmT, unsigned short* __restrict__ Wc) {
  const int idx = blockIdx.x * blockDim.x + threadIdx.x;   // 1M entries
  const int n = idx >> 10, k = idx & 1023;
  float s = 0.f;
#pragma unroll
  for (int r = 0; r < RANK; ++r) s += A3[k * RANK + r] * BmT[n * RANK + r];
  Wc[idx] = f2bf(wb[idx] + ALPHA * s);
}

// ---------------------------------------------------------------------------
// Kernel 3: stream-convert x f32 -> bf16 (memory bound; VALU cost hidden).
// ---------------------------------------------------------------------------
__global__ void convert_x(const float* __restrict__ X, unsigned short* __restrict__ Xb) {
  const size_t idx = ((size_t)blockIdx.x * blockDim.x + threadIdx.x) * 8;
  v4f f0 = *(const v4f*)(X + idx);
  v4f f1 = *(const v4f*)(X + idx + 4);
  v8u u;
#pragma unroll
  for (int e = 0; e < 4; ++e) { u[e] = f2bf(f0[e]); u[4 + e] = f2bf(f1[e]); }
  *(v8u*)(Xb + idx) = u;
}

// ---------------------------------------------------------------------------
// Kernel 4a: fused GEMM, A pre-converted to bf16.
// Wave tile: 32(M) x 64(N) = 2x4 accumulators; 8 WMMA + 12 b128 loads / K-step.
// ---------------------------------------------------------------------------
__global__ void __launch_bounds__(256)
tt_fused_gemm_bf16(const unsigned short* __restrict__ Xb, const unsigned short* __restrict__ Wc,
                   const float* __restrict__ bias, float* __restrict__ Out, int M) {
  const int K = D_IN, N = D_OUT;
  const int lane = threadIdx.x & 31;
  const int wave = threadIdx.x >> 5;
  const int w = blockIdx.x * 8 + wave;
  const int mTiles = M >> 5;
  const int mb = w % mTiles, nb = w / mTiles;
  const int m0 = mb << 5, n0 = nb << 6;
  const int lm = lane & 15;
  const int half = lane >> 4;

  const v8f vz = {0.f, 0.f, 0.f, 0.f, 0.f, 0.f, 0.f, 0.f};
  v8f acc[2][4];
#pragma unroll
  for (int i = 0; i < 2; ++i)
#pragma unroll
    for (int j = 0; j < 4; ++j) acc[i][j] = vz;

  // A 16-bit 16x32 layout: lanes 0-15 K in {0..7,16..23}, lanes 16-31 {8..15,24..31}
  const unsigned short* ap[2];
  ap[0] = Xb + (size_t)(m0 + lm) * K + (half << 3);
  ap[1] = Xb + (size_t)(m0 + 16 + lm) * K + (half << 3);
  // B: lane = column n, same K layout as A
  const unsigned short* bp[4];
#pragma unroll
  for (int j = 0; j < 4; ++j)
    bp[j] = Wc + (size_t)(n0 + j * 16 + lm) * K + (half << 3);

  for (int k0 = 0; k0 < K; k0 += 32) {
    if (k0 + 64 < K) __builtin_prefetch(ap[0] + k0 + 64, 0, 0);

    v16bf a[2], b[4];
#pragma unroll
    for (int i = 0; i < 2; ++i) {
      const unsigned short* q = ap[i] + k0;
      v8u lo = *(const v8u*)(q);           // K = k0+8h    .. k0+8h+7
      v8u hi = *(const v8u*)(q + 16);      // K = k0+16+8h .. k0+23+8h
      a[i] = __builtin_bit_cast(v16bf, cat8(lo, hi));
    }
#pragma unroll
    for (int j = 0; j < 4; ++j) {
      const unsigned short* q = bp[j] + k0;
      v8u lo = *(const v8u*)(q);           // K = k0+8h    .. k0+8h+7
      v8u hi = *(const v8u*)(q + 16);      // K = k0+16+8h .. k0+23+8h
      b[j] = __builtin_bit_cast(v16bf, cat8(lo, hi));
    }
#pragma unroll
    for (int i = 0; i < 2; ++i)
#pragma unroll
      for (int j = 0; j < 4; ++j)
        acc[i][j] = __builtin_amdgcn_wmma_f32_16x16x32_bf16(
            false, a[i], false, b[j], (short)0, acc[i][j], false, false);
  }

#pragma unroll
  for (int j = 0; j < 4; ++j) {
    const int n = n0 + j * 16 + lm;
    const float bv = bias[n];
#pragma unroll
    for (int i = 0; i < 2; ++i) {
      const int rbase = m0 + (i << 4) + (half << 3);
#pragma unroll
      for (int r = 0; r < 8; ++r)
        Out[(size_t)(rbase + r) * N + n] = acc[i][j][r] + bv;
    }
  }
}

// ---------------------------------------------------------------------------
// Kernel 4b: fallback GEMM converting A on the fly (used if ws too small).
// ---------------------------------------------------------------------------
__global__ void __launch_bounds__(256)
tt_fused_gemm_f32a(const float* __restrict__ X, const unsigned short* __restrict__ Wc,
                   const float* __restrict__ bias, float* __restrict__ Out, int M) {
  const int K = D_IN, N = D_OUT;
  const int lane = threadIdx.x & 31;
  const int wave = threadIdx.x >> 5;
  const int w = blockIdx.x * 8 + wave;
  const int mTiles = M >> 5;
  const int mb = w % mTiles, nb = w / mTiles;
  const int m0 = mb << 5, n0 = nb << 6;
  const int lm = lane & 15;
  const int half = lane >> 4;

  const v8f vz = {0.f, 0.f, 0.f, 0.f, 0.f, 0.f, 0.f, 0.f};
  v8f acc[2][4];
#pragma unroll
  for (int i = 0; i < 2; ++i)
#pragma unroll
    for (int j = 0; j < 4; ++j) acc[i][j] = vz;

  const float* ap[2];
  ap[0] = X + (size_t)(m0 + lm) * K;
  ap[1] = X + (size_t)(m0 + 16 + lm) * K;
  const int aoff = half << 3;
  const unsigned short* bp[4];
#pragma unroll
  for (int j = 0; j < 4; ++j)
    bp[j] = Wc + (size_t)(n0 + j * 16 + lm) * K + (half << 3);

  for (int k0 = 0; k0 < K; k0 += 32) {
    if (k0 + 64 < K) __builtin_prefetch(ap[0] + k0 + 64, 0, 0);

    v16bf a[2], b[4];
#pragma unroll
    for (int i = 0; i < 2; ++i) {
      const float* p = ap[i] + k0 + aoff;
      v4f f0 = *(const v4f*)(p);
      v4f f1 = *(const v4f*)(p + 4);
      v4f f2 = *(const v4f*)(p + 16);
      v4f f3 = *(const v4f*)(p + 20);
      v16u u;
#pragma unroll
      for (int e = 0; e < 4; ++e) {
        u[e]      = f2bf(f0[e]);
        u[4 + e]  = f2bf(f1[e]);
        u[8 + e]  = f2bf(f2[e]);
        u[12 + e] = f2bf(f3[e]);
      }
      a[i] = __builtin_bit_cast(v16bf, u);
    }
#pragma unroll
    for (int j = 0; j < 4; ++j) {
      const unsigned short* q = bp[j] + k0;
      v8u lo = *(const v8u*)(q);
      v8u hi = *(const v8u*)(q + 16);
      b[j] = __builtin_bit_cast(v16bf, cat8(lo, hi));
    }
#pragma unroll
    for (int i = 0; i < 2; ++i)
#pragma unroll
      for (int j = 0; j < 4; ++j)
        acc[i][j] = __builtin_amdgcn_wmma_f32_16x16x32_bf16(
            false, a[i], false, b[j], (short)0, acc[i][j], false, false);
  }

#pragma unroll
  for (int j = 0; j < 4; ++j) {
    const int n = n0 + j * 16 + lm;
    const float bv = bias[n];
#pragma unroll
    for (int i = 0; i < 2; ++i) {
      const int rbase = m0 + (i << 4) + (half << 3);
#pragma unroll
      for (int r = 0; r < 8; ++r)
        Out[(size_t)(rbase + r) * N + n] = acc[i][j][r] + bv;
    }
  }
}

extern "C" void kernel_launch(void* const* d_in, const int* in_sizes, int n_in,
                              void* d_out, int out_size, void* d_ws, size_t ws_size,
                              hipStream_t stream) {
  const float* x      = (const float*)d_in[0];
  const float* w_base = (const float*)d_in[1];
  const float* b_base = (const float*)d_in[2];
  const float* c0 = (const float*)d_in[3];
  const float* c1 = (const float*)d_in[4];
  const float* c2 = (const float*)d_in[5];
  const float* c3 = (const float*)d_in[6];
  const float* c4 = (const float*)d_in[7];
  const float* c5 = (const float*)d_in[8];
  float* out = (float*)d_out;

  const int M = in_sizes[0] / D_IN;                 // 16384 rows

  // Workspace: Wc bf16 (2 MB) | A3 (32 KB) | BmT (32 KB) | optional Xb bf16 (32 MB)
  char* ws = (char*)d_ws;
  unsigned short* Wc = (unsigned short*)ws;
  const size_t wcBytes   = (size_t)D_OUT * D_IN * sizeof(unsigned short);
  float* A3  = (float*)(ws + wcBytes);
  float* BmT = A3 + (size_t)D_IN * RANK;
  const size_t fixedBytes = wcBytes + (size_t)(D_IN + D_OUT) * RANK * sizeof(float);
  const size_t xbBytes    = (size_t)M * D_IN * sizeof(unsigned short);
  unsigned short* Xb = (unsigned short*)(ws + fixedBytes);

  const int waves  = (M >> 5) * (D_OUT >> 6);
  const int blocks = waves >> 3;

  tt_contract_cores<<<1, 256, 0, stream>>>(c0, c1, c2, c3, c4, c5, A3, BmT);
  build_fused_w<<<(D_IN * D_OUT) / 256, 256, 0, stream>>>(w_base, A3, BmT, Wc);

  if (ws_size >= fixedBytes + xbBytes) {
    const int cvtBlocks = (int)(((size_t)M * D_IN) / (8 * 256));
    convert_x<<<cvtBlocks, 256, 0, stream>>>(x, Xb);
    tt_fused_gemm_bf16<<<blocks, 256, 0, stream>>>(Xb, Wc, b_base, out, M);
  } else {
    tt_fused_gemm_f32a<<<blocks, 256, 0, stream>>>(x, Wc, b_base, out, M);
  }
}
